// NodeGNN_77309411328100
// MI455X (gfx1250) — matrix operand order repaired and checked by reading, not compiled, not run on hardware
//
#include <hip/hip_runtime.h>

#define NND   100000
#define NED   1600000
#define HDIM  128
#define NCLS  8
#define BNEPS 1e-5f
#define SLOPE 0.01f

typedef __attribute__((ext_vector_type(16))) _Float16 v16h;
typedef __attribute__((ext_vector_type(8)))  float    v8f;

static __device__ __forceinline__ void atomAddF(float* p, float v) {
    __hip_atomic_fetch_add(p, v, __ATOMIC_RELAXED, __HIP_MEMORY_SCOPE_AGENT);
}

// leaky relu without compares: for SLOPE<1, max(y, SLOPE*y) == lrelu(y)
static __device__ __forceinline__ float lrelu(float y) {
    return fmaxf(y, SLOPE * y);
}

// ---- degree + norm ---------------------------------------------------------
__global__ void deg_kernel(const int* __restrict__ src, const int* __restrict__ dst,
                           float* __restrict__ degO, float* __restrict__ degI, int nE) {
    int e = blockIdx.x * blockDim.x + threadIdx.x;
    if (e >= nE) return;
    atomAddF(degO + src[e], 1.0f);
    atomAddF(degI + dst[e], 1.0f);
}

__global__ void norm_kernel(float* __restrict__ degO, float* __restrict__ degI, int n) {
    int i = blockIdx.x * blockDim.x + threadIdx.x;
    if (i >= n) return;
    degO[i] = rsqrtf(fmaxf(degO[i], 1.0f));
    degI[i] = rsqrtf(fmaxf(degI[i], 1.0f));
}

// ---- edge scatter: agg[dst] += x[src] * out_norm[src] ----------------------
// one wave (32 lanes) per edge, float4 per lane => 128 features
__global__ __launch_bounds__(256) void scatter_kernel(
    const float* __restrict__ x, const int* __restrict__ src, const int* __restrict__ dst,
    const float* __restrict__ onorm, float* __restrict__ agg, int nE)
{
    long long t = (long long)blockIdx.x * blockDim.x + threadIdx.x;
    int e    = (int)(t >> 5);
    int lane = (int)(t & 31);
    if (e >= nE) return;
    int s = src[e], d = dst[e];
    float w = onorm[s];
    float4 v = ((const float4*)(x + (long long)s * HDIM))[lane];
    float* ap = agg + (long long)d * HDIM + lane * 4;
    atomAddF(ap + 0, v.x * w);
    atomAddF(ap + 1, v.y * w);
    atomAddF(ap + 2, v.z * w);
    atomAddF(ap + 3, v.w * w);
}

// ---- convert (agg * in_norm) -> f16 A matrix -------------------------------
__global__ void convert_a_kernel(const float* __restrict__ agg, const float* __restrict__ innorm,
                                 _Float16* __restrict__ a16, int total) {
    int i = blockIdx.x * blockDim.x + threadIdx.x;
    if (i >= total) return;
    a16[i] = (_Float16)(agg[i] * innorm[i >> 7]);
}

// ---- convert + transpose W (k-major [128,128]) -> f16 column-major ---------
__global__ void convert_w_kernel(const float* __restrict__ W, _Float16* __restrict__ Wt) {
    int i = blockIdx.x * blockDim.x + threadIdx.x;
    if (i >= HDIM * HDIM) return;
    int k = i >> 7, j = i & 127;
    Wt[j * HDIM + k] = (_Float16)W[i];
}

// ---- WMMA GEMM: C[N,128] = A16[N,128] @ W + bias (compile-time lrelu) ------
// block = 256 threads = 8 waves; block covers a 16-row band, wave w -> cols 16w..16w+15
template <int ACT>
__global__ __launch_bounds__(256) void gemm_kernel(
    const _Float16* __restrict__ A, const _Float16* __restrict__ Bt,
    const float* __restrict__ bias, float* __restrict__ C)
{
    const int lane = threadIdx.x & 31;
    const int wave = threadIdx.x >> 5;
    const int hl   = lane >> 4;      // lane half (0: lanes 0-15, 1: lanes 16-31)
    const int lm   = lane & 15;
    const int m0   = blockIdx.x * 16;
    const int col  = wave * 16 + lm;

    const _Float16* arow = A  + (long long)(m0 + lm) * HDIM;
    const _Float16* bcol = Bt + (long long)col * HDIM;

    v8f c = {};
#pragma unroll
    for (int kb = 0; kb < HDIM; kb += 32) {
        v16h a, b;
        // A 16x32 fragment (ISA 7.12.2): half 0 -> K {0..7,16..23}, half 1 -> K {8..15,24..31}
#pragma unroll
        for (int t = 0; t < 8; ++t) {
            a[t]     = arow[kb + 8 * hl + t];
            a[8 + t] = arow[kb + 16 + 8 * hl + t];
        }
        // B 32x16 fragment: lane holds one column; half 0 -> K {0..7,16..23}, half 1 -> K {8..15,24..31}
#pragma unroll
        for (int t = 0; t < 8; ++t) {
            b[t]     = bcol[kb + 8 * hl + t];
            b[8 + t] = bcol[kb + 16 + 8 * hl + t];
        }
        c = __builtin_amdgcn_wmma_f32_16x16x32_f16(false, a, false, b, (short)0, c,
                                                   false, false);
    }

    float bb = bias[col];
#pragma unroll
    for (int v = 0; v < 8; ++v) {     // VGPR v holds rows m0+v (half 0) / m0+v+8 (half 1)
        float y = c[v] + bb;
        if (ACT) y = lrelu(y);
        C[(long long)(m0 + v + 8 * hl) * HDIM + col] = y;
    }
}

// ---- batchnorm stats: per-feature sum / sumsq ------------------------------
__global__ __launch_bounds__(128) void bn_stats_kernel(const float* __restrict__ h,
                                                       float* __restrict__ sums,
                                                       float* __restrict__ sumsq, int nRows) {
    int f  = threadIdx.x;
    int r0 = blockIdx.x * 64;
    int r1 = r0 + 64; if (r1 > nRows) r1 = nRows;
    float s = 0.0f, s2 = 0.0f;
    for (int r = r0; r < r1; ++r) {
        float v = h[(long long)r * HDIM + f];
        s += v;
        s2 = fmaf(v, v, s2);
    }
    atomAddF(sums + f, s);
    atomAddF(sumsq + f, s2);
}

// ---- batchnorm apply + leaky relu (in place) -------------------------------
__global__ void bn_apply_kernel(float* __restrict__ h, const float* __restrict__ sums,
                                const float* __restrict__ sumsq,
                                const float* __restrict__ gamma, const float* __restrict__ beta,
                                int total, float invN) {
    int i = blockIdx.x * blockDim.x + threadIdx.x;
    if (i >= total) return;
    int f = i & 127;
    float mean = sums[f] * invN;
    float var  = fmaf(-mean, mean, sumsq[f] * invN);
    float sc   = gamma[f] * rsqrtf(var + BNEPS);
    h[i] = lrelu((h[i] - mean) * sc + beta[f]);
}

// ---- classifier: out[N,8] = hn @ Wc + bc -----------------------------------
__global__ void cls_kernel(const float* __restrict__ hn, const float* __restrict__ Wc,
                           const float* __restrict__ bc, float* __restrict__ out, int total) {
    int t = blockIdx.x * blockDim.x + threadIdx.x;
    if (t >= total) return;
    int node = t >> 3, c = t & 7;
    const float* row = hn + (long long)node * HDIM;
    float s = bc[c];
#pragma unroll 8
    for (int k = 0; k < HDIM; ++k) s = fmaf(row[k], Wc[k * NCLS + c], s);
    out[t] = s;
}

// ---------------------------------------------------------------------------
template <int ACT>
static void run_layer(const float* x, const int* src, const int* dst,
                      const float* onorm, const float* innorm,
                      const float* W, const float* bias,
                      float* buf /* agg in, output out */, _Float16* a16, _Float16* w16,
                      hipStream_t stream)
{
    hipMemsetAsync(buf, 0, (size_t)NND * HDIM * sizeof(float), stream);
    {
        long long thr = (long long)NED * 32;
        scatter_kernel<<<(unsigned)((thr + 255) / 256), 256, 0, stream>>>(x, src, dst, onorm, buf, NED);
    }
    convert_a_kernel<<<(NND * HDIM + 255) / 256, 256, 0, stream>>>(buf, innorm, a16, NND * HDIM);
    convert_w_kernel<<<(HDIM * HDIM + 255) / 256, 256, 0, stream>>>(W, w16);
    gemm_kernel<ACT><<<NND / 16, 256, 0, stream>>>(a16, w16, bias, buf);
}

extern "C" void kernel_launch(void* const* d_in, const int* in_sizes, int n_in,
                              void* d_out, int out_size, void* d_ws, size_t ws_size,
                              hipStream_t stream) {
    (void)in_sizes; (void)n_in; (void)out_size; (void)ws_size;
    const float* node_feat = (const float*)d_in[0];
    const int*   src       = (const int*)d_in[1];
    const int*   dst       = (const int*)d_in[2];
    const float* W1 = (const float*)d_in[3];  const float* b1 = (const float*)d_in[4];
    const float* W2 = (const float*)d_in[5];  const float* b2 = (const float*)d_in[6];
    const float* g1 = (const float*)d_in[7];  const float* be1 = (const float*)d_in[8];
    const float* g2 = (const float*)d_in[9];  const float* be2 = (const float*)d_in[10];
    const float* Wn1 = (const float*)d_in[11]; const float* bn1 = (const float*)d_in[12];
    const float* Wn2 = (const float*)d_in[13]; const float* bn2 = (const float*)d_in[14];
    const float* Wc  = (const float*)d_in[15]; const float* bc  = (const float*)d_in[16];
    float* out = (float*)d_out;

    // workspace layout (16B aligned offsets)
    char* ws = (char*)d_ws;
    float*    onorm  = (float*)(ws + 0);                // N floats (deg -> norm in place)
    float*    innorm = (float*)(ws + 400000);           // N floats
    float*    sums   = (float*)(ws + 800000);           // 128
    float*    sumsq  = (float*)(ws + 800512);           // 128
    _Float16* w16    = (_Float16*)(ws + 801024);        // 128*128
    _Float16* a16    = (_Float16*)(ws + 833792);        // N*128
    float*    B0     = (float*)(ws + 26433792);         // N*128
    float*    B1     = (float*)(ws + 77633792);         // N*128

    const float invN = 1.0f / (float)NND;

    // degrees -> inverse-sqrt norms
    hipMemsetAsync(onorm, 0, 2 * (size_t)NND * sizeof(float), stream); // onorm+innorm contiguous
    deg_kernel<<<(NED + 255) / 256, 256, 0, stream>>>(src, dst, onorm, innorm, NED);
    norm_kernel<<<(NND + 255) / 256, 256, 0, stream>>>(onorm, innorm, NND);

    // conv1 + bn1 + lrelu
    run_layer<0>(node_feat, src, dst, onorm, innorm, W1, b1, B0, a16, w16, stream);
    hipMemsetAsync(sums, 0, 1024, stream);
    bn_stats_kernel<<<(NND + 63) / 64, 128, 0, stream>>>(B0, sums, sumsq, NND);
    bn_apply_kernel<<<(NND * HDIM + 255) / 256, 256, 0, stream>>>(B0, sums, sumsq, g1, be1,
                                                                  NND * HDIM, invN);
    // conv2 + bn2 + lrelu
    run_layer<0>(B0, src, dst, onorm, innorm, W2, b2, B1, a16, w16, stream);
    hipMemsetAsync(sums, 0, 1024, stream);
    bn_stats_kernel<<<(NND + 63) / 64, 128, 0, stream>>>(B1, sums, sumsq, NND);
    bn_apply_kernel<<<(NND * HDIM + 255) / 256, 256, 0, stream>>>(B1, sums, sumsq, g2, be2,
                                                                  NND * HDIM, invN);
    // convnode1 (lrelu fused into gemm epilogue)
    run_layer<1>(B1, src, dst, onorm, innorm, Wn1, bn1, B0, a16, w16, stream);
    // convnode2
    run_layer<1>(B0, src, dst, onorm, innorm, Wn2, bn2, B1, a16, w16, stream);

    // classifier
    cls_kernel<<<(NND * NCLS + 255) / 256, 256, 0, stream>>>(B1, Wc, bc, out, NND * NCLS);
}
